// HAttentionLayer_20186346291762
// MI455X (gfx1250) — matrix operand order repaired and checked by reading, not compiled, not run on hardware
//
#include <hip/hip_runtime.h>
#include <math.h>

typedef __attribute__((ext_vector_type(16))) _Float16 v16h;
typedef __attribute__((ext_vector_type(8)))  float    v8f;
typedef __attribute__((ext_vector_type(4)))  unsigned v4u;
typedef __attribute__((ext_vector_type(8)))  unsigned v8u;

#define T_LEN 2048
#define D_DIM 1024
#define H_NUM 16
#define S_DIM 64
#define HD_DIM 64
#define LV_NUM 12

// ---------------- workspace layout (bytes) ----------------
#define OFF_XH   ((size_t)0)                                  // T*D f16   4 MB
#define OFF_WQH  (OFF_XH  + (size_t)T_LEN*D_DIM*2)            // D*D f16   2 MB
#define OFF_WKH  (OFF_WQH + (size_t)D_DIM*D_DIM*2)
#define OFF_WVH  (OFF_WKH + (size_t)D_DIM*D_DIM*2)
#define OFF_WOH  (OFF_WVH + (size_t)D_DIM*D_DIM*2)
#define OFF_QH   (OFF_WOH + (size_t)D_DIM*D_DIM*2)            // T*D f16
#define OFF_KH   (OFF_QH  + (size_t)T_LEN*D_DIM*2)
#define OFF_VH   (OFF_KH  + (size_t)T_LEN*D_DIM*2)
#define OFF_YH   (OFF_VH  + (size_t)T_LEN*D_DIM*2)
#define OFF_O    (OFF_YH  + (size_t)T_LEN*D_DIM*2)            // T*D f32   8 MB
#define OFF_G    (OFF_O   + (size_t)T_LEN*D_DIM*4)            // T*H f32
#define OFF_CG   (OFF_G   + (size_t)T_LEN*H_NUM*4)
#define OFF_LAM  (OFF_CG  + (size_t)T_LEN*H_NUM*4)            // T*H*LV f32

// ---------------- fp32 -> fp16 convert ----------------
__global__ void cvt_f16_kernel(const float* __restrict__ s, _Float16* __restrict__ d, int n) {
    int i = blockIdx.x * blockDim.x + threadIdx.x;
    if (i < n) d[i] = (_Float16)s[i];
}

// ---------------- small projections: g = A*softplus(x@Wdt^T+b), Lam = softplus(Lp*(x@Wdl^T)) ----------------
__global__ void proj_small_kernel(const float* __restrict__ x, const float* __restrict__ Wdt,
                                  const float* __restrict__ Wdl, const float* __restrict__ A_log,
                                  const float* __restrict__ dt_bias, const float* __restrict__ Lp,
                                  float* __restrict__ gout, float* __restrict__ LamOut) {
    int t = blockIdx.x;
    __shared__ float xs[D_DIM];
    for (int i = threadIdx.x; i < D_DIM; i += blockDim.x) xs[i] = x[(size_t)t * D_DIM + i];
    __syncthreads();
    int o = threadIdx.x;
    if (o < H_NUM) {
        const float* wr = Wdt + (size_t)o * D_DIM;
        float acc = 0.f;
        for (int dd = 0; dd < D_DIM; ++dd) acc += xs[dd] * wr[dd];
        float z = acc + dt_bias[o];
        float sp = (z > 20.f) ? z : log1pf(__expf(z));
        gout[(size_t)t * H_NUM + o] = -__expf(A_log[o]) * sp;
    } else if (o < H_NUM + H_NUM * LV_NUM) {
        int q = o - H_NUM;
        const float* wr = Wdl + (size_t)q * D_DIM;
        float acc = 0.f;
        for (int dd = 0; dd < D_DIM; ++dd) acc += xs[dd] * wr[dd];
        float z = Lp[q] * acc;   // Lp is [H,LV] flat, q = h*LV+l
        LamOut[(size_t)t * (H_NUM * LV_NUM) + q] = (z > 20.f) ? z : log1pf(__expf(z));
    }
}

// ---------------- per-head inclusive cumsum of g over time (wave scan) ----------------
__global__ void scan_g_kernel(const float* __restrict__ gin, float* __restrict__ cg) {
    int h = blockIdx.x;
    int lane = threadIdx.x;
    float base = 0.f;
    for (int t0 = 0; t0 < T_LEN; t0 += 32) {
        float v = gin[(size_t)(t0 + lane) * H_NUM + h];
        #pragma unroll
        for (int dlt = 1; dlt < 32; dlt <<= 1) {
            float u = __shfl_up(v, (unsigned)dlt, 32);
            if (lane >= dlt) v += u;
        }
        cg[(size_t)(t0 + lane) * H_NUM + h] = v + base;
        base += __shfl(v, 31, 32);
    }
}

// ---------------- WMMA GEMM: C[t,n] = sum_k A[t,k]*Bw[n,k]  (per-wave 16x64 tile) ----------------
// mode 0: store f16 into Ch. mode 1: store f32 into Cf with residual add.
__global__ __launch_bounds__(128) void gemm_wmma_kernel(const _Float16* __restrict__ A,
                                                        const _Float16* __restrict__ Bw,
                                                        _Float16* __restrict__ Ch,
                                                        float* __restrict__ Cf,
                                                        const float* __restrict__ resid,
                                                        int mode) {
    const int Ktot = D_DIM, Ntot = D_DIM;
    int lane = threadIdx.x & 31;
    int wave = threadIdx.x >> 5;
    int lg = lane >> 4, ln = lane & 15;
    int t0 = blockIdx.y * 16;
    int n0 = blockIdx.x * 256 + wave * 64;

    const v8f vzero = {0.f, 0.f, 0.f, 0.f, 0.f, 0.f, 0.f, 0.f};
    v8f acc[4] = {vzero, vzero, vzero, vzero};
    const _Float16* arow = A + (size_t)(t0 + ln) * Ktot;

    for (int k0 = 0; k0 < Ktot; k0 += 32) {
        union { uint4 q[2]; v16h v; } ua;
        ua.q[0] = *(const uint4*)(arow + k0 + 16 * lg);
        ua.q[1] = *(const uint4*)(arow + k0 + 16 * lg + 8);
        #pragma unroll
        for (int j = 0; j < 4; ++j) {
            const _Float16* brow = Bw + (size_t)(n0 + j * 16 + ln) * Ktot + k0 + 16 * lg;
            union { uint4 q[2]; v16h v; } ub;
            ub.q[0] = *(const uint4*)(brow);
            ub.q[1] = *(const uint4*)(brow + 8);
            acc[j] = __builtin_amdgcn_wmma_f32_16x16x32_f16(false, ua.v, false, ub.v,
                                                            (short)0, acc[j], false, false);
        }
    }
    #pragma unroll
    for (int j = 0; j < 4; ++j) {
        #pragma unroll
        for (int r = 0; r < 8; ++r) {
            size_t idx = (size_t)(t0 + r + 8 * lg) * Ntot + n0 + j * 16 + ln;
            if (mode == 0) Ch[idx] = (_Float16)acc[j][r];
            else           Cf[idx] = acc[j][r] + resid[idx];
        }
    }
}

// ---------------- fused causal hierarchical attention (flash-style, per-wave 16 query rows) ----------------
// V tiles are DMA'd into LDS by the Tensor Data Mover; consumption gated on TENSORcnt.
__global__ __launch_bounds__(128) void attn_kernel(const _Float16* __restrict__ Qh,
                                                   const _Float16* __restrict__ Kh,
                                                   const _Float16* __restrict__ Vh,
                                                   const float* __restrict__ cg,
                                                   const float* __restrict__ Lam,
                                                   _Float16* __restrict__ Yh) {
    __shared__ __align__(16) _Float16 VL[4][32][64];   // V tile per wave (TDM destination)
    __shared__ __align__(16) _Float16 PL[4][16][32];   // gated probs per wave
    __shared__ float LL[4][16][LV_NUM];                // lambda per query row

    int lane = threadIdx.x & 31;
    int w = threadIdx.x >> 5;
    int lg = lane >> 4, ln = lane & 15;
    int h = blockIdx.y;
    int t0 = blockIdx.x * 64 + w * 16;

    // lambda tile for the 16 query rows
    for (int k = lane; k < 16 * LV_NUM; k += 32)
        LL[w][k / LV_NUM][k % LV_NUM] =
            Lam[(size_t)(t0 + k / LV_NUM) * (H_NUM * LV_NUM) + h * LV_NUM + (k % LV_NUM)];

    // Q A-fragments (K = 0..31 and 32..63 of head_dim)
    const _Float16* qrow = Qh + (size_t)(t0 + ln) * D_DIM + h * 64;
    union { uint4 q[2]; v16h v; } uq0, uq1;
    uq0.q[0] = *(const uint4*)(qrow + 16 * lg);
    uq0.q[1] = *(const uint4*)(qrow + 16 * lg + 8);
    uq1.q[0] = *(const uint4*)(qrow + 32 + 16 * lg);
    uq1.q[1] = *(const uint4*)(qrow + 32 + 16 * lg + 8);

    float cgt[8];
    #pragma unroll
    for (int r = 0; r < 8; ++r) cgt[r] = cg[(size_t)(t0 + r + 8 * lg) * H_NUM + h];

    const v8f vzero = {0.f, 0.f, 0.f, 0.f, 0.f, 0.f, 0.f, 0.f};
    v8f accY[4] = {vzero, vzero, vzero, vzero};

    // TDM descriptor pieces (wave-uniform). Group1: data_size=2B, tile 64x32,
    // tensor dims 64x32, dim0 stride 1024 elements. Groups 2/3 zero (2D tensor).
    unsigned ldsVL = (unsigned)(unsigned long long)(&VL[w][0][0]);
    v8u g1 = { 0x00010000u,          // data_size=1 (2 bytes), no flags
               64u << 16,            // tensor_dim0 = 64 (bits 79:48, low part)
               32u << 16,            // tensor_dim1 = 32 (bits 111:80, low part)
               64u << 16,            // tile_dim0 = 64 (bits 127:112)
               32u,                  // tile_dim1 = 32 (bits 143:128)
               1024u,                // tensor_dim0_stride = 1024 (bits 207:160)
               0u, 0u };
    v4u gz = { 0u, 0u, 0u, 0u };

    asm volatile("s_wait_dscnt 0" ::: "memory");   // LL visible to whole wave

    int niter = (t0 + 47) >> 5;                    // covers all s <= t0+15
    for (int it = 0; it < niter; ++it) {
        int s0 = it * 32;
        // ---- issue TDM: V[s0..s0+31][h*64..h*64+63] -> VL[w] ----
        asm volatile("s_wait_dscnt 0" ::: "memory");   // prior-iter LDS readers done (TDM unordered vs DS)
        {
            unsigned long long gaddr =
                (unsigned long long)(const void*)(Vh + (size_t)s0 * D_DIM + h * 64);
            v4u g0;
            g0.x = 1u;                                     // count=1, no gather
            g0.y = ldsVL;                                  // lds_addr
            g0.z = (unsigned)(gaddr & 0xffffffffu);        // global_addr[31:0]
            g0.w = (unsigned)((gaddr >> 32) & 0x1ffffffu) | (2u << 30);  // addr[56:32] | type=2
            asm volatile("tensor_load_to_lds %0, %1, %2, %3"
                         :: "s"(g0), "s"(g1), "s"(gz), "s"(gz)
                         : "memory");
        }
        // prefetch next s-tile of K and V while this tile computes
        if (it + 1 < niter) {
            const _Float16* kn = Kh + (size_t)(s0 + 32 + lane) * D_DIM + h * 64;
            const _Float16* vn = Vh + (size_t)(s0 + 32 + lane) * D_DIM + h * 64;
            __builtin_prefetch((const void*)kn, 0, 3);
            __builtin_prefetch((const void*)vn, 0, 3);
        }
        // ---- scores + decay/lambda gate, two 16-key subtiles ----
        #pragma unroll
        for (int sub = 0; sub < 2; ++sub) {
            int sg = s0 + sub * 16 + ln;           // this lane's key index (N)
            const _Float16* krow = Kh + (size_t)sg * D_DIM + h * 64;
            union { uint4 q[2]; v16h v; } uk0, uk1;
            uk0.q[0] = *(const uint4*)(krow + 16 * lg);
            uk0.q[1] = *(const uint4*)(krow + 16 * lg + 8);
            uk1.q[0] = *(const uint4*)(krow + 32 + 16 * lg);
            uk1.q[1] = *(const uint4*)(krow + 32 + 16 * lg + 8);
            v8f sc = vzero;
            sc = __builtin_amdgcn_wmma_f32_16x16x32_f16(false, uq0.v, false, uk0.v,
                                                        (short)0, sc, false, false);
            sc = __builtin_amdgcn_wmma_f32_16x16x32_f16(false, uq1.v, false, uk1.v,
                                                        (short)0, sc, false, false);
            float cgs = cg[(size_t)sg * H_NUM + h];
            #pragma unroll
            for (int r = 0; r < 8; ++r) {
                int trow = t0 + r + 8 * lg;
                float wv = 0.f;
                if (sg <= trow) {
                    int xr = trow ^ sg;
                    int lv = xr ? (32 - __builtin_clz((unsigned)xr)) : 0;  // Fenwick level
                    float lam = LL[w][r + 8 * lg][lv];
                    wv = sc[r] * 0.125f * __expf(cgt[r] - cgs) * lam;
                }
                PL[w][r + 8 * lg][sub * 16 + ln] = (_Float16)wv;
            }
        }
        asm volatile("" ::: "memory");
        asm volatile("s_wait_dscnt 0" ::: "memory");  // P tile readable cross-lane
        __builtin_amdgcn_s_wait_tensorcnt(0);         // V tile landed in LDS
        // P as A-fragment (transpose via LDS)
        union { uint4 q[2]; v16h v; } up;
        up.q[0] = *(const uint4*)&PL[w][ln][16 * lg];
        up.q[1] = *(const uint4*)&PL[w][ln][16 * lg + 8];
        // Y += P(16x32) * V(32x64)  (4 N-tiles)
        #pragma unroll
        for (int j = 0; j < 4; ++j) {
            v16h bv;
            #pragma unroll
            for (int i = 0; i < 16; ++i)
                bv[i] = VL[w][16 * lg + i][j * 16 + ln];
            accY[j] = __builtin_amdgcn_wmma_f32_16x16x32_f16(false, up.v, false, bv,
                                                             (short)0, accY[j], false, false);
        }
        asm volatile("" ::: "memory");
    }
    #pragma unroll
    for (int j = 0; j < 4; ++j)
        #pragma unroll
        for (int r = 0; r < 8; ++r)
            Yh[(size_t)(t0 + r + 8 * lg) * D_DIM + h * 64 + j * 16 + ln] = (_Float16)accY[j][r];
}

// ---------------- LayerNorm over D ----------------
__global__ void ln_kernel(const float* __restrict__ o, const float* __restrict__ gw,
                          const float* __restrict__ bw, float* __restrict__ out) {
    int t = blockIdx.x;
    const float* row = o + (size_t)t * D_DIM;
    float s = 0.f, s2 = 0.f;
    for (int i = threadIdx.x; i < D_DIM; i += 256) { float v = row[i]; s += v; s2 += v * v; }
    #pragma unroll
    for (int dlt = 16; dlt > 0; dlt >>= 1) {
        s  += __shfl_down(s,  (unsigned)dlt, 32);
        s2 += __shfl_down(s2, (unsigned)dlt, 32);
    }
    __shared__ float shs[8], shs2[8], stat[2];
    int lane = threadIdx.x & 31, wid = threadIdx.x >> 5;
    if (lane == 0) { shs[wid] = s; shs2[wid] = s2; }
    __syncthreads();
    if (threadIdx.x == 0) {
        float a = 0.f, b = 0.f;
        for (int i = 0; i < 8; ++i) { a += shs[i]; b += shs2[i]; }
        float mu = a / (float)D_DIM;
        stat[0] = mu;
        stat[1] = rsqrtf(b / (float)D_DIM - mu * mu + 1e-5f);
    }
    __syncthreads();
    float mu = stat[0], inv = stat[1];
    for (int i = threadIdx.x; i < D_DIM; i += 256)
        out[(size_t)t * D_DIM + i] = (row[i] - mu) * inv * gw[i] + bw[i];
}

extern "C" void kernel_launch(void* const* d_in, const int* in_sizes, int n_in,
                              void* d_out, int out_size, void* d_ws, size_t ws_size,
                              hipStream_t stream) {
    (void)in_sizes; (void)n_in; (void)out_size; (void)ws_size;
    const float* x     = (const float*)d_in[0];
    const float* Wq    = (const float*)d_in[1];
    const float* Wk    = (const float*)d_in[2];
    const float* Wv    = (const float*)d_in[3];
    const float* Wdl   = (const float*)d_in[4];
    const float* Wdt   = (const float*)d_in[5];
    const float* Wout  = (const float*)d_in[6];
    const float* A_log = (const float*)d_in[7];
    const float* dt_b  = (const float*)d_in[8];
    const float* Lp    = (const float*)d_in[9];
    const float* ln_g  = (const float*)d_in[10];
    const float* ln_b  = (const float*)d_in[11];

    char* ws = (char*)d_ws;
    _Float16* xh  = (_Float16*)(ws + OFF_XH);
    _Float16* wqh = (_Float16*)(ws + OFF_WQH);
    _Float16* wkh = (_Float16*)(ws + OFF_WKH);
    _Float16* wvh = (_Float16*)(ws + OFF_WVH);
    _Float16* woh = (_Float16*)(ws + OFF_WOH);
    _Float16* Qh  = (_Float16*)(ws + OFF_QH);
    _Float16* Kh  = (_Float16*)(ws + OFF_KH);
    _Float16* Vh  = (_Float16*)(ws + OFF_VH);
    _Float16* Yh  = (_Float16*)(ws + OFF_YH);
    float*    ob  = (float*)(ws + OFF_O);
    float*    gb  = (float*)(ws + OFF_G);
    float*    cgb = (float*)(ws + OFF_CG);
    float*    lmb = (float*)(ws + OFF_LAM);

    const int nTD = T_LEN * D_DIM;
    const int nDD = D_DIM * D_DIM;
    cvt_f16_kernel<<<(nTD + 255) / 256, 256, 0, stream>>>(x, xh, nTD);
    cvt_f16_kernel<<<(nDD + 255) / 256, 256, 0, stream>>>(Wq, wqh, nDD);
    cvt_f16_kernel<<<(nDD + 255) / 256, 256, 0, stream>>>(Wk, wkh, nDD);
    cvt_f16_kernel<<<(nDD + 255) / 256, 256, 0, stream>>>(Wv, wvh, nDD);
    cvt_f16_kernel<<<(nDD + 255) / 256, 256, 0, stream>>>(Wout, woh, nDD);

    proj_small_kernel<<<T_LEN, 256, 0, stream>>>(x, Wdt, Wdl, A_log, dt_b, Lp, gb, lmb);
    scan_g_kernel<<<H_NUM, 32, 0, stream>>>(gb, cgb);

    dim3 ggrid(4, T_LEN / 16);
    gemm_wmma_kernel<<<ggrid, 128, 0, stream>>>(xh, wqh, Qh, nullptr, nullptr, 0);
    gemm_wmma_kernel<<<ggrid, 128, 0, stream>>>(xh, wkh, Kh, nullptr, nullptr, 0);
    gemm_wmma_kernel<<<ggrid, 128, 0, stream>>>(xh, wvh, Vh, nullptr, nullptr, 0);

    attn_kernel<<<dim3(T_LEN / 64, H_NUM), 128, 0, stream>>>(Qh, Kh, Vh, cgb, lmb, Yh);

    gemm_wmma_kernel<<<ggrid, 128, 0, stream>>>(Yh, woh, nullptr, ob, x, 1);
    ln_kernel<<<T_LEN, 256, 0, stream>>>(ob, ln_g, ln_b, (float*)d_out);
}
